// MambaLayer_69380901700087
// MI455X (gfx1250) — matrix operand order repaired and checked by reading, not compiled, not run on hardware
//
#include <hip/hip_runtime.h>

typedef __bf16 bf16;
typedef __attribute__((ext_vector_type(8)))  __bf16 bf16x8;
typedef __attribute__((ext_vector_type(16))) __bf16 bf16x16;
typedef __attribute__((ext_vector_type(8)))  float  f32x8;

// ---- model dims ----
static constexpr int Dm  = 1024;   // d_model
static constexpr int EDm = 2048;   // d_inner
static constexpr int Kc  = 4;      // d_conv
static constexpr int Nst = 16;     // d_state
static constexpr int Rr  = 64;     // dt_rank
static constexpr int Ee  = 8;      // experts
static constexpr int FFm = 2048;   // mlp size
static constexpr int Tt  = 2048;   // B*L tokens
static constexpr int XP  = Rr + 2 * Nst;  // 96

__device__ __forceinline__ float softplus_f(float x) {
  return x > 0.f ? x + log1pf(expf(-x)) : log1pf(expf(x));
}
__device__ __forceinline__ float silu_f(float x) {
  return x / (1.f + expf(-x));
}

// ---------------- generic f32 -> bf16 convert ----------------
__global__ __launch_bounds__(256) void cvt_bf16_kernel(const float* __restrict__ s,
                                                       bf16* __restrict__ d, long n) {
  long i = (long)blockIdx.x * 256 + threadIdx.x;
  if (i < n) d[i] = (bf16)s[i];
}

// ---------------- rmsnorm (one block per token row) ----------------
__global__ __launch_bounds__(256)
void rmsnorm_kernel(const float* __restrict__ x, const float* __restrict__ w,
                    float* __restrict__ of32, bf16* __restrict__ obf) {
  __shared__ float red[8];
  const int row = blockIdx.x, tid = threadIdx.x;
  const float* xr = x + (long)row * Dm;
  float s = 0.f;
  for (int i = tid; i < Dm; i += 256) { float v = xr[i]; s += v * v; }
  #pragma unroll
  for (int o = 16; o > 0; o >>= 1) s += __shfl_down(s, o, 32);
  if ((tid & 31) == 0) red[tid >> 5] = s;
  __syncthreads();
  if (tid == 0) {
    float t = 0.f;
    #pragma unroll
    for (int i = 0; i < 8; ++i) t += red[i];
    red[0] = rsqrtf(t / (float)Dm + 1e-6f);
  }
  __syncthreads();
  const float sc = red[0];
  for (int i = tid; i < Dm; i += 256) {
    float v = xr[i] * sc * w[i];
    if (of32) of32[(long)row * Dm + i] = v;
    obf[(long)row * Dm + i] = (bf16)v;
  }
}

// ---------------- depthwise causal conv (K=4) + bias + silu ----------------
__global__ __launch_bounds__(256)
void conv_kernel(const float* __restrict__ xz, const float* __restrict__ cw,
                 const float* __restrict__ cb, float* __restrict__ xc,
                 bf16* __restrict__ xcbf) {
  long idx = (long)blockIdx.x * 256 + threadIdx.x;   // t*ED + e
  if (idx >= (long)Tt * EDm) return;
  int e = (int)(idx & (EDm - 1));
  long t = idx >> 11;
  float acc = cb[e];
  #pragma unroll
  for (int j = 0; j < Kc; ++j) {
    long tt = t - (Kc - 1) + j;
    if (tt >= 0) acc += cw[e * Kc + j] * xz[tt * (2 * EDm) + e];
  }
  float s = silu_f(acc);
  xc[idx] = s;
  xcbf[idx] = (bf16)s;
}

// ---------------- extract dt columns (first 64 of xproj) to bf16 ----------------
__global__ __launch_bounds__(256)
void dt_extract_kernel(const float* __restrict__ xproj, bf16* __restrict__ dtbf) {
  int i = blockIdx.x * 256 + threadIdx.x;  // Tt*Rr
  if (i >= Tt * Rr) return;
  int t = i >> 6, c = i & 63;
  dtbf[i] = (bf16)xproj[t * XP + c];
}

// ---------------- selective scan: lane = (channel, state) ----------------
__global__ __launch_bounds__(256)
void scan_kernel(const float* __restrict__ delta, const float* __restrict__ xc,
                 const float* __restrict__ xz, const float* __restrict__ xproj,
                 const float* __restrict__ A_log, const float* __restrict__ D_skip,
                 bf16* __restrict__ yact_bf) {
  const int tid = threadIdx.x;
  const int eloc = tid >> 4, n = tid & 15;
  const int e = blockIdx.x * 16 + eloc;
  const float Av = -__expf(A_log[e * Nst + n]);
  const float dsk = D_skip[e];
  float h = 0.f;
  __shared__ float sD[16], sX[16], sZ[16], sB[16], sC[16];
  for (int t = 0; t < Tt; ++t) {
    __syncthreads();
    if (tid < 16) {
      int ee = blockIdx.x * 16 + tid;
      sD[tid] = delta[(long)t * EDm + ee];
      sX[tid] = xc[(long)t * EDm + ee];
      sZ[tid] = xz[(long)t * (2 * EDm) + EDm + ee];
    } else if (tid < 32) {
      sB[tid - 16] = xproj[t * XP + Rr + (tid - 16)];
    } else if (tid < 48) {
      sC[tid - 32] = xproj[t * XP + Rr + Nst + (tid - 32)];
    }
    __syncthreads();
    float d = sD[eloc], xv = sX[eloc];
    float dA = __expf(d * Av);
    h = dA * h + d * sB[n] * xv;
    float part = h * sC[n];
    part += __shfl_xor(part, 8, 16);
    part += __shfl_xor(part, 4, 16);
    part += __shfl_xor(part, 2, 16);
    part += __shfl_xor(part, 1, 16);
    if (n == 0) {
      float y = part + dsk * xv;
      float z = sZ[eloc];
      y *= silu_f(z);
      yact_bf[(long)t * EDm + e] = (bf16)y;
    }
  }
}

// ---------------- router: wave per token, softmax + top-2 ----------------
__global__ __launch_bounds__(256)
void router_kernel(const float* __restrict__ h2, const float* __restrict__ Wr,
                   float* __restrict__ logits_out, int* __restrict__ selA,
                   int* __restrict__ selB, float* __restrict__ wA,
                   float* __restrict__ wB, int* __restrict__ counts) {
  const int wid = threadIdx.x >> 5, lane = threadIdx.x & 31;
  const int t = blockIdx.x * 8 + wid;
  if (t >= Tt) return;
  float acc[Ee];
  #pragma unroll
  for (int e = 0; e < Ee; ++e) acc[e] = 0.f;
  const float* xr = h2 + (long)t * Dm;
  for (int d = lane; d < Dm; d += 32) {
    float xv = xr[d];
    #pragma unroll
    for (int e = 0; e < Ee; ++e) acc[e] += xv * Wr[e * Dm + d];
  }
  #pragma unroll
  for (int o = 16; o > 0; o >>= 1)
    #pragma unroll
    for (int e = 0; e < Ee; ++e) acc[e] += __shfl_down(acc[e], o, 32);
  if (lane == 0) {
    float mx = acc[0];
    #pragma unroll
    for (int e = 1; e < Ee; ++e) mx = fmaxf(mx, acc[e]);
    float p[Ee], sum = 0.f;
    #pragma unroll
    for (int e = 0; e < Ee; ++e) { p[e] = expf(acc[e] - mx); sum += p[e]; }
    #pragma unroll
    for (int e = 0; e < Ee; ++e) p[e] /= sum;
    int s0 = 0; float v0 = p[0];
    #pragma unroll
    for (int e = 1; e < Ee; ++e) if (p[e] > v0) { v0 = p[e]; s0 = e; }
    int s1 = -1; float v1 = -1.f;
    #pragma unroll
    for (int e = 0; e < Ee; ++e) if (e != s0 && p[e] > v1) { v1 = p[e]; s1 = e; }
    #pragma unroll
    for (int e = 0; e < Ee; ++e) logits_out[t * Ee + e] = acc[e];
    selA[t] = s0; selB[t] = s1; wA[t] = v0; wB[t] = v1;
    atomicAdd(&counts[s0], 1);
    atomicAdd(&counts[s1], 1);
  }
}

__global__ void zero_counts_kernel(int* __restrict__ p) {
  if (threadIdx.x < Ee) p[threadIdx.x] = 0;
}

__global__ void offsets_kernel(const int* __restrict__ counts, int* __restrict__ offs,
                               int* __restrict__ cursor) {
  if (threadIdx.x == 0) {
    int a = 0;
    for (int e = 0; e < Ee; ++e) { offs[e] = a; a += counts[e]; cursor[e] = 0; }
    offs[Ee] = a;
  }
}

__global__ __launch_bounds__(256)
void assign_kernel(const int* __restrict__ selA, const int* __restrict__ selB,
                   const float* __restrict__ wA, const float* __restrict__ wB,
                   const int* __restrict__ offs, int* __restrict__ cursor,
                   int* __restrict__ rowSrc, int* __restrict__ rowDst,
                   float* __restrict__ rowScale) {
  int t = blockIdx.x * 256 + threadIdx.x;
  if (t >= Tt) return;
  int e0 = selA[t];
  int r0 = offs[e0] + atomicAdd(&cursor[e0], 1);
  rowSrc[r0] = t; rowDst[r0] = t * 2; rowScale[r0] = wA[t];
  int e1 = selB[t];
  int r1 = offs[e1] + atomicAdd(&cursor[e1], 1);
  rowSrc[r1] = t; rowDst[r1] = t * 2 + 1; rowScale[r1] = wB[t];
}

// ---------------- silu(g)*u -> bf16 ----------------
__global__ __launch_bounds__(256)
void silumul_kernel(const float* __restrict__ g, const float* __restrict__ u,
                    bf16* __restrict__ hm, long n) {
  long i = (long)blockIdx.x * 256 + threadIdx.x;
  if (i < n) hm[i] = (bf16)(silu_f(g[i]) * u[i]);
}

// ---------------- final combine: out = x2 + ye[slot0] + ye[slot1] ----------------
__global__ __launch_bounds__(256)
void combine_kernel(const float* __restrict__ x2, const float* __restrict__ ye,
                    float* __restrict__ out) {
  long i = (long)blockIdx.x * 256 + threadIdx.x;
  if (i >= (long)Tt * Dm) return;
  long t = i >> 10;
  int d = (int)(i & (Dm - 1));
  out[i] = x2[i] + ye[(t * 2) * Dm + d] + ye[(t * 2 + 1) * Dm + d];
}

// ---------------- WMMA bf16 GEMM: C = act(A*B^T + bias)*scale + Cadd ----------------
// A: [M,K] bf16 row-major (optionally row-gathered), B: [N,K] bf16 row-major.
// Block tile 128x64, 8 waves as 4x2, wave tile 32x32 (4 WMMA accumulators).
// Double-buffered LDS (one barrier per k-step); branchless inner loop:
// out-of-range rows/cols are clamped on load and masked only at store.
__global__ __launch_bounds__(256)
void gemm_bf16_kernel(const bf16* __restrict__ A, const bf16* __restrict__ Bmat,
                      float* __restrict__ Cout, int M, int N, int K, long strideBz,
                      const int* __restrict__ cnts, const int* __restrict__ offs,
                      const int* __restrict__ rowSrc, const int* __restrict__ rowDst,
                      const float* __restrict__ rowScale, const float* __restrict__ bias,
                      const float* __restrict__ Cadd, int ldAdd, int act) {
  const int z = blockIdx.z;
  const int Meff = cnts ? cnts[z] : M;
  const int base = offs ? offs[z] : 0;
  const int m0 = blockIdx.y * 128;
  if (m0 >= Meff) return;
  const int n0 = blockIdx.x * 64;
  const bf16* Bz = Bmat + (long)z * strideBz;

  __shared__ __align__(16) bf16 sA[2][128][40];
  __shared__ __align__(16) bf16 sB[2][64][40];

  const int tid = threadIdx.x;
  const int lane = tid & 31;
  const int wid = tid >> 5;
  const int wm = (wid & 3) * 32;     // wave m offset in tile (0,32,64,96)
  const int wn = (wid >> 2) * 32;    // wave n offset in tile (0,32)
  const int fr = lane & 15;
  const int kloA = (lane >> 4) * 8;  // A frag: K = kloA..+7, kloA+16..+23
  const int kloB = (lane >> 4) * 8;  // B frag: K = kloB..+7, kloB+16..+23

  // staging assignment: 2 A chunks + 1 B chunk per thread, row fixed across k
  const int ar0 = tid >> 2,         ac0 = (tid & 3) * 8;
  const int ar1 = (tid + 256) >> 2, ac1 = (tid & 3) * 8;
  const int br  = tid >> 2,         bc  = (tid & 3) * 8;

  // clamped global row pointers (garbage rows masked at store)
  const bf16* aP0;
  const bf16* aP1;
  {
    int l0 = base + min(m0 + ar0, Meff - 1);
    int l1 = base + min(m0 + ar1, Meff - 1);
    int g0 = rowSrc ? rowSrc[l0] : l0;
    int g1 = rowSrc ? rowSrc[l1] : l1;
    aP0 = A + (long)g0 * K + ac0;
    aP1 = A + (long)g1 * K + ac1;
  }
  const bf16* bP = Bz + (long)min(n0 + br, N - 1) * K + bc;

  f32x8 acc[2][2];
  #pragma unroll
  for (int mi = 0; mi < 2; ++mi)
    #pragma unroll
    for (int ni = 0; ni < 2; ++ni)
      #pragma unroll
      for (int i = 0; i < 8; ++i) acc[mi][ni][i] = 0.f;

  // prologue: stage k=0
  bf16x8 ra0 = *(const bf16x8*)(aP0);
  bf16x8 ra1 = *(const bf16x8*)(aP1);
  bf16x8 rb  = *(const bf16x8*)(bP);
  *(bf16x8*)(&sA[0][ar0][ac0]) = ra0;
  *(bf16x8*)(&sA[0][ar1][ac1]) = ra1;
  *(bf16x8*)(&sB[0][br][bc])   = rb;
  __syncthreads();

  int p = 0;
  for (int k0 = 0; k0 < K; k0 += 32) {
    const int nk = k0 + 32;
    const bool more = nk < K;
    if (more) {           // prefetch next tile while computing current
      ra0 = *(const bf16x8*)(aP0 + nk);
      ra1 = *(const bf16x8*)(aP1 + nk);
      rb  = *(const bf16x8*)(bP + nk);
    }

    union { bf16x16 v; bf16x8 h[2]; } fa0, fa1, fb0, fb1;
    fa0.h[0] = *(const bf16x8*)(&sA[p][wm + fr][kloA]);
    fa0.h[1] = *(const bf16x8*)(&sA[p][wm + fr][kloA + 16]);
    fa1.h[0] = *(const bf16x8*)(&sA[p][wm + 16 + fr][kloA]);
    fa1.h[1] = *(const bf16x8*)(&sA[p][wm + 16 + fr][kloA + 16]);
    fb0.h[0] = *(const bf16x8*)(&sB[p][wn + fr][kloB]);
    fb0.h[1] = *(const bf16x8*)(&sB[p][wn + fr][kloB + 16]);
    fb1.h[0] = *(const bf16x8*)(&sB[p][wn + 16 + fr][kloB]);
    fb1.h[1] = *(const bf16x8*)(&sB[p][wn + 16 + fr][kloB + 16]);

    acc[0][0] = __builtin_amdgcn_wmma_f32_16x16x32_bf16(false, fa0.v, false, fb0.v,
                                                        (short)0, acc[0][0], false, false);
    acc[0][1] = __builtin_amdgcn_wmma_f32_16x16x32_bf16(false, fa0.v, false, fb1.v,
                                                        (short)0, acc[0][1], false, false);
    acc[1][0] = __builtin_amdgcn_wmma_f32_16x16x32_bf16(false, fa1.v, false, fb0.v,
                                                        (short)0, acc[1][0], false, false);
    acc[1][1] = __builtin_amdgcn_wmma_f32_16x16x32_bf16(false, fa1.v, false, fb1.v,
                                                        (short)0, acc[1][1], false, false);

    if (more) {
      const int q = p ^ 1;
      *(bf16x8*)(&sA[q][ar0][ac0]) = ra0;
      *(bf16x8*)(&sA[q][ar1][ac1]) = ra1;
      *(bf16x8*)(&sB[q][br][bc])   = rb;
      __syncthreads();
      p = q;
    }
  }

  // C/D layout: VGPR v -> M = v + 8*(lane>=16); N = lane&15
  const int rbase = (lane >> 4) * 8;
  #pragma unroll
  for (int mi = 0; mi < 2; ++mi) {
    #pragma unroll
    for (int v = 0; v < 8; ++v) {
      int ml = m0 + wm + mi * 16 + rbase + v;
      if (ml >= Meff) continue;
      int lrow = base + ml;
      int drow = rowDst ? rowDst[lrow] : lrow;
      float scl = rowScale ? rowScale[lrow] : 1.f;
      #pragma unroll
      for (int ni = 0; ni < 2; ++ni) {
        int gc = n0 + wn + ni * 16 + fr;
        if (gc >= N) continue;
        float x = acc[mi][ni][v];
        if (bias) x += bias[gc];
        if (act == 1) x = softplus_f(x);
        x *= scl;
        if (Cadd) x += Cadd[(long)drow * ldAdd + gc];
        Cout[(long)drow * N + gc] = x;
      }
    }
  }
}

// =======================================================================
static inline void launch_gemm(hipStream_t s, const bf16* A, const bf16* B, float* C,
                               int M, int N, int K, long strideBz, int Z,
                               const int* cnts, const int* offs, const int* rowSrc,
                               const int* rowDst, const float* rowScale,
                               const float* bias, const float* Cadd, int ldAdd, int act) {
  dim3 grid((N + 63) / 64, (M + 127) / 128, Z);
  gemm_bf16_kernel<<<grid, 256, 0, s>>>(A, B, C, M, N, K, strideBz, cnts, offs,
                                        rowSrc, rowDst, rowScale, bias, Cadd, ldAdd, act);
}

static inline void launch_cvt(hipStream_t s, const float* src, bf16* dst, long n) {
  cvt_bf16_kernel<<<(int)((n + 255) / 256), 256, 0, s>>>(src, dst, n);
}

extern "C" void kernel_launch(void* const* d_in, const int* in_sizes, int n_in,
                              void* d_out, int out_size, void* d_ws, size_t ws_size,
                              hipStream_t stream) {
  (void)in_sizes; (void)n_in; (void)out_size; (void)ws_size;
  const float* x       = (const float*)d_in[0];
  const float* rms1_w  = (const float*)d_in[1];
  const float* rms2_w  = (const float*)d_in[2];
  const float* W_in    = (const float*)d_in[3];
  const float* conv_w  = (const float*)d_in[4];
  const float* conv_b  = (const float*)d_in[5];
  const float* W_xproj = (const float*)d_in[6];
  const float* W_dt    = (const float*)d_in[7];
  const float* b_dt    = (const float*)d_in[8];
  const float* A_log   = (const float*)d_in[9];
  const float* D_skip  = (const float*)d_in[10];
  const float* W_out   = (const float*)d_in[11];
  const float* W_rout  = (const float*)d_in[12];
  const float* gate_w  = (const float*)d_in[13];
  const float* up_w    = (const float*)d_in[14];
  const float* down_w  = (const float*)d_in[15];

  float* out_x      = (float*)d_out;                       // [T,D]
  float* out_logits = (float*)d_out + (long)Tt * Dm;       // [T,E]

  // ---- workspace layout ----
  char* w = (char*)d_ws;
  auto alloc = [&](size_t bytes) -> void* {
    void* p = (void*)w;
    w += (bytes + 255) & ~(size_t)255;
    return p;
  };
  bf16*  h1_bf    = (bf16*) alloc((size_t)Tt * Dm * 2);
  float* xz       = (float*)alloc((size_t)Tt * 2 * EDm * 4);
  float* xc       = (float*)alloc((size_t)Tt * EDm * 4);
  bf16*  xc_bf    = (bf16*) alloc((size_t)Tt * EDm * 2);
  float* xproj    = (float*)alloc((size_t)Tt * XP * 4);
  bf16*  dt_bf    = (bf16*) alloc((size_t)Tt * Rr * 2);
  float* delta    = (float*)alloc((size_t)Tt * EDm * 4);
  bf16*  yact_bf  = (bf16*) alloc((size_t)Tt * EDm * 2);
  float* x2       = (float*)alloc((size_t)Tt * Dm * 4);
  float* h2_f     = (float*)alloc((size_t)Tt * Dm * 4);
  bf16*  h2_bf    = (bf16*) alloc((size_t)Tt * Dm * 2);
  float* gbuf     = (float*)alloc((size_t)2 * Tt * FFm * 4);
  float* ubuf     = (float*)alloc((size_t)2 * Tt * FFm * 4);
  bf16*  hmid_bf  = (bf16*) alloc((size_t)2 * Tt * FFm * 2);
  float* ye       = (float*)alloc((size_t)2 * Tt * Dm * 4);
  bf16*  W_in_bf  = (bf16*) alloc((size_t)2 * EDm * Dm * 2);
  bf16*  W_xp_bf  = (bf16*) alloc((size_t)XP * EDm * 2);
  bf16*  W_dt_bf  = (bf16*) alloc((size_t)EDm * Rr * 2);
  bf16*  W_out_bf = (bf16*) alloc((size_t)Dm * EDm * 2);
  bf16*  gate_bf  = (bf16*) alloc((size_t)Ee * FFm * Dm * 2);
  bf16*  up_bf    = (bf16*) alloc((size_t)Ee * FFm * Dm * 2);
  bf16*  down_bf  = (bf16*) alloc((size_t)Ee * Dm * FFm * 2);
  int*   counts   = (int*)  alloc(Ee * 4);
  int*   cursor   = (int*)  alloc(Ee * 4);
  int*   offs     = (int*)  alloc((Ee + 1) * 4);
  int*   rowSrc   = (int*)  alloc((size_t)2 * Tt * 4);
  int*   rowDst   = (int*)  alloc((size_t)2 * Tt * 4);
  float* rowScale = (float*)alloc((size_t)2 * Tt * 4);
  int*   selA     = (int*)  alloc((size_t)Tt * 4);
  int*   selB     = (int*)  alloc((size_t)Tt * 4);
  float* wA       = (float*)alloc((size_t)Tt * 4);
  float* wB       = (float*)alloc((size_t)Tt * 4);

  // ---- weight conversions to bf16 (deterministic, every call) ----
  launch_cvt(stream, W_in,    W_in_bf,  (long)2 * EDm * Dm);
  launch_cvt(stream, W_xproj, W_xp_bf,  (long)XP * EDm);
  launch_cvt(stream, W_dt,    W_dt_bf,  (long)EDm * Rr);
  launch_cvt(stream, W_out,   W_out_bf, (long)Dm * EDm);
  launch_cvt(stream, gate_w,  gate_bf,  (long)Ee * FFm * Dm);
  launch_cvt(stream, up_w,    up_bf,    (long)Ee * FFm * Dm);
  launch_cvt(stream, down_w,  down_bf,  (long)Ee * Dm * FFm);

  zero_counts_kernel<<<1, 32, 0, stream>>>(counts);

  // ---- Mamba branch ----
  rmsnorm_kernel<<<Tt, 256, 0, stream>>>(x, rms1_w, nullptr, h1_bf);
  // xz = h1 @ W_in^T      [T, 2*ED]
  launch_gemm(stream, h1_bf, W_in_bf, xz, Tt, 2 * EDm, Dm, 0, 1,
              nullptr, nullptr, nullptr, nullptr, nullptr, nullptr, nullptr, 0, 0);
  conv_kernel<<<(Tt * EDm) / 256, 256, 0, stream>>>(xz, conv_w, conv_b, xc, xc_bf);
  // xproj = xc @ W_xproj^T   [T, 96]
  launch_gemm(stream, xc_bf, W_xp_bf, xproj, Tt, XP, EDm, 0, 1,
              nullptr, nullptr, nullptr, nullptr, nullptr, nullptr, nullptr, 0, 0);
  dt_extract_kernel<<<(Tt * Rr) / 256, 256, 0, stream>>>(xproj, dt_bf);
  // delta = softplus(dt @ W_dt^T + b_dt)   [T, ED]
  launch_gemm(stream, dt_bf, W_dt_bf, delta, Tt, EDm, Rr, 0, 1,
              nullptr, nullptr, nullptr, nullptr, nullptr, b_dt, nullptr, 0, 1);
  scan_kernel<<<EDm / 16, 256, 0, stream>>>(delta, xc, xz, xproj, A_log, D_skip, yact_bf);
  // x2 = yact @ W_out^T + x   [T, D]
  launch_gemm(stream, yact_bf, W_out_bf, x2, Tt, Dm, EDm, 0, 1,
              nullptr, nullptr, nullptr, nullptr, nullptr, nullptr, x, Dm, 0);

  // ---- MoE branch ----
  rmsnorm_kernel<<<Tt, 256, 0, stream>>>(x2, rms2_w, h2_f, h2_bf);
  router_kernel<<<Tt / 8, 256, 0, stream>>>(h2_f, W_rout, out_logits,
                                            selA, selB, wA, wB, counts);
  offsets_kernel<<<1, 32, 0, stream>>>(counts, offs, cursor);
  assign_kernel<<<Tt / 256, 256, 0, stream>>>(selA, selB, wA, wB, offs, cursor,
                                              rowSrc, rowDst, rowScale);
  // gate / up: gathered rows of h2 against per-expert weights
  launch_gemm(stream, h2_bf, gate_bf, gbuf, Tt, FFm, Dm, (long)FFm * Dm, Ee,
              counts, offs, rowSrc, nullptr, nullptr, nullptr, nullptr, 0, 0);
  launch_gemm(stream, h2_bf, up_bf, ubuf, Tt, FFm, Dm, (long)FFm * Dm, Ee,
              counts, offs, rowSrc, nullptr, nullptr, nullptr, nullptr, 0, 0);
  silumul_kernel<<<(int)(((long)2 * Tt * FFm + 255) / 256), 256, 0, stream>>>(
      gbuf, ubuf, hmid_bf, (long)2 * Tt * FFm);
  // down: scaled by routing weight, scattered to ye[(t,slot)]
  launch_gemm(stream, hmid_bf, down_bf, ye, Tt, Dm, FFm, (long)Dm * FFm, Ee,
              counts, offs, nullptr, rowDst, rowScale, nullptr, nullptr, 0, 0);
  combine_kernel<<<(int)(((long)Tt * Dm + 255) / 256), 256, 0, stream>>>(x2, ye, out_x);
}
